// GlobalFusion_30176440222534
// MI455X (gfx1250) — matrix operand order repaired and checked by reading, not compiled, not run on hardware
//
#include <hip/hip_runtime.h>
#include <stdint.h>

// ---------------------------------------------------------------------------
// MI455X (gfx1250, wave32) implementation.
// GEMM-heavy parts (1x1 convs, minGRU projections, up-projection) run on
// v_wmma_f32_16x16x32_bf16 (bf16 inputs, f32 accumulate). The GEMM A-tile is
// staged into LDS by the Tensor Data Mover (tensor_load_to_lds) with double
// buffering, synchronized with s_wait_tensorcnt. B-fragments for a K-step are
// prefetched into a register array so the WMMAs issue back-to-back.
// Memory-bound parts (transpose/pack, depthwise 3x3, 6-step GRU scans,
// softmax view-pool, pixel-shuffle + final 3x3 conv) are plain VALU kernels.
// Note: the reference's flip is on the H axis (faithful port of
// torch.flip(dims=[1])) which is a no-op for a scan along V, so both GRUs are
// forward scans along V. jax.image.resize((512,512)->(512,512)) is identity.
// ---------------------------------------------------------------------------

#define USE_TDM 1

typedef __attribute__((ext_vector_type(16))) __bf16 v16bf;
typedef __attribute__((ext_vector_type(8)))  float  v8f;
typedef __attribute__((ext_vector_type(4)))  unsigned int v4u;
typedef __attribute__((ext_vector_type(8)))  int    v8i;
typedef __attribute__((ext_vector_type(4)))  int    v4i;

#define Bn   2
#define Vn   6
#define Cn   128
#define Hn   128
#define Wn   128
#define HWn  (Hn * Wn)        // 16384
#define BVn  (Bn * Vn)        // 12
#define NPIX (BVn * HWn)      // 196608
#define BHW  (Bn * HWn)       // 32768

__device__ __forceinline__ float gelu_exact(float x) {
  return 0.5f * x * (1.0f + erff(x * 0.70710678118654752f));
}
__device__ __forceinline__ float sigmoid_(float x) {
  return 1.0f / (1.0f + __expf(-x));
}

#if USE_TDM
// ---------------------------------------------------------------------------
// Issue one TDM 2-D tile load: rows x 16 dwords (64B) from a row-major matrix
// with row stride row_stride_dw (dwords) into LDS at lds_off (packed tile).
// D# bit packing per CDNA5 ISA ch.8 (group0: count/lds_addr/global_addr/type,
// group1: data_size=4B, tensor dims, tile dims, dim0 stride). Groups 2/3 NULL.
// ---------------------------------------------------------------------------
__device__ __forceinline__ void tdm_load_tile(const void* gptr, uint32_t lds_off,
                                              uint32_t row_stride_dw, uint32_t rows) {
  const uint64_t ga = (uint64_t)(uintptr_t)gptr;
  v4u g0;
  g0[0] = 1u;                                   // count=1 (valid), no gather
  g0[1] = lds_off;                              // LDS byte address
  g0[2] = (uint32_t)ga;                         // global_addr[31:0]
  g0[3] = ((uint32_t)(ga >> 32) & 0x01FFFFFFu)  // global_addr[56:32]
          | 0x80000000u;                        // type=2 ("image")
  const uint32_t tdim0 = row_stride_dw;         // tensor dim0 (dwords)
  const uint32_t tdim1 = 0x00100000u;           // generous row count (OOB only)
  v8i g1;
  g1[0] = (int)0x00020000u;                                   // data_size=4B
  g1[1] = (int)((tdim0 & 0xFFFFu) << 16);                     // tensor_dim0 lo
  g1[2] = (int)(((tdim0 >> 16) & 0xFFFFu) |
                ((tdim1 & 0xFFFFu) << 16));                   // dim0 hi | dim1 lo
  g1[3] = (int)(((tdim1 >> 16) & 0xFFFFu) | (16u << 16));     // dim1 hi | tile_dim0=16
  g1[4] = (int)(rows & 0xFFFFu);                              // tile_dim1=rows, tile_dim2=0
  g1[5] = (int)row_stride_dw;                                 // tensor_dim0_stride lo
  g1[6] = 0;                                                  // stride0 hi | stride1 lo
  g1[7] = 0;                                                  // stride1 hi
  const v4i z4 = {0, 0, 0, 0};
#if __clang_major__ >= 23
  const v8i z8 = {0, 0, 0, 0, 0, 0, 0, 0};
  __builtin_amdgcn_tensor_load_to_lds(g0, g1, z4, z4, z8, 0);
#else
  __builtin_amdgcn_tensor_load_to_lds(g0, g1, z4, z4, 0);
#endif
}
#endif  // USE_TDM

// ---------------------------------------------------------------------------
// Pack NCHW f32 activations -> pixel-major bf16 (n*ldst + col_off + c),
// LDS-tiled 32x32 transpose so both global read and write are coalesced.
// grid (HW/32, C/32, BV), block (32,8)
// ---------------------------------------------------------------------------
__global__ void pack_act(const float* __restrict__ src, __bf16* __restrict__ dst,
                         int col_off, int ldst) {
  __shared__ float tile[32][33];
  const int bv = blockIdx.z;
  const int c0 = blockIdx.y * 32;
  const int p0 = blockIdx.x * 32;
#pragma unroll
  for (int i = 0; i < 4; ++i) {
    const int c = c0 + threadIdx.y + i * 8;
    tile[threadIdx.y + i * 8][threadIdx.x] =
        src[((size_t)bv * Cn + c) * HWn + p0 + threadIdx.x];
  }
  __syncthreads();
#pragma unroll
  for (int i = 0; i < 4; ++i) {
    const int p = p0 + threadIdx.y + i * 8;
    dst[((size_t)bv * HWn + p) * ldst + col_off + c0 + threadIdx.x] =
        (__bf16)tile[threadIdx.x][threadIdx.y + i * 8];
  }
}

// ---------------------------------------------------------------------------
// Pack a row-major f32 weight Wt[N rows][K cols] (GEMM computes X @ Wt^T) into
// the per-lane WMMA B-fragment layout for 16x16x32 bf16:
//   B 32x16 tile: lanes 0-15 hold K={0..7,16..23}, lanes 16-31 hold K={8..15,24..31};
//   column n = lane&15. Stored so each lane reads 16 contiguous bf16 (32B).
// ---------------------------------------------------------------------------
__global__ void pack_weight(const float* __restrict__ Wt, __bf16* __restrict__ Bsw,
                            int K, int N) {
  const int t = blockIdx.x * blockDim.x + threadIdx.x;
  if (t >= K * N) return;
  const int e    = t & 15;
  const int lane = (t >> 4) & 31;
  const int frag = t >> 9;            // kt * (N/16) + nt
  const int NT   = N >> 4;
  const int nt   = frag % NT;
  const int kt   = frag / NT;
  const int n    = nt * 16 + (lane & 15);
  const int k    = kt * 32 + ((lane < 16) ? 0 : 8) + ((e < 8) ? e : (8 + e));
  Bsw[t] = (__bf16)Wt[(size_t)n * K + k];
}

// ---------------------------------------------------------------------------
// WMMA GEMM: out(M,N) = epi( A(M,K)bf16 @ Bsw + bias [+ residual(NCHW f32)] )
// block = 256 threads = 8 wave32; block tile = 128 rows x N.
// Each wave owns a 16-row slice and all N/16 column tiles. Per K-step:
// A tile (128x32 bf16 = 8KB) arrives via double-buffered TDM (wave 0 issues
// tensor_load(kt+1) right after the barrier so the DMA overlaps compute),
// all N/16 B-fragments load into a register array (one clause, overlapped
// latency), then the WMMAs issue back-to-back.
// ---------------------------------------------------------------------------
template <int K_, int N_, int EPI /*0 none,1 gelu*/, bool OBF16, bool RES, bool BIAS>
__global__ void wmma_gemm(const __bf16* __restrict__ A,
                          const __bf16* __restrict__ Bsw,
                          const float* __restrict__ bias,
                          const float* __restrict__ res_nchw,
                          void* __restrict__ outv) {
  constexpr int NT = N_ / 16;
  constexpr int KT = K_ / 32;
  __shared__ __bf16 Atile[2][128 * 32];      // double-buffered 8KB tiles
  const int tid  = threadIdx.x;
  const int wave = tid >> 5;
  const int lane = tid & 31;
  const int rowBase = blockIdx.x * 128;

  v8f acc[NT];
#pragma unroll
  for (int j = 0; j < NT; ++j) acc[j] = {};

#if USE_TDM
  if (wave == 0)
    tdm_load_tile(A + (size_t)rowBase * K_, (uint32_t)(uintptr_t)&Atile[0][0],
                  K_ / 2, 128);
#endif

  for (int kt = 0; kt < KT; ++kt) {
#if USE_TDM
    if (wave == 0) __builtin_amdgcn_s_wait_tensorcnt(0);
    __syncthreads();                          // publish buffer kt&1 to all waves
    if (wave == 0 && kt + 1 < KT)             // prefetch next tile (other buffer)
      tdm_load_tile(A + (size_t)rowBase * K_ + (kt + 1) * 32,
                    (uint32_t)(uintptr_t)&Atile[(kt + 1) & 1][0], K_ / 2, 128);
    const __bf16* As = &Atile[kt & 1][0];
#else
    __syncthreads();
    {   // cooperative stage: 128 rows x 16 dwords = 2048 dwords, 8 per thread
      const uint32_t* A32 = (const uint32_t*)A;
      uint32_t* S32 = (uint32_t*)&Atile[0][0];
#pragma unroll
      for (int i = 0; i < 8; ++i) {
        const int j = tid + 256 * i;
        const int r = j >> 4, cu = j & 15;
        S32[j] = A32[(size_t)(rowBase + r) * (K_ / 2) + kt * 16 + cu];
      }
    }
    __syncthreads();
    const __bf16* As = &Atile[0][0];
#endif
    // A fragment (16x32 bf16): lane<16 -> K chunks {0..7,16..23}, else {8..15,24..31}
    const int arow = wave * 16 + (lane & 15);
    const int k1   = (lane < 16) ? 0 : 8;
    v16bf afrag;
#pragma unroll
    for (int i = 0; i < 8; ++i) afrag[i] = As[arow * 32 + k1 + i];
#pragma unroll
    for (int i = 0; i < 8; ++i) afrag[8 + i] = As[arow * 32 + k1 + 16 + i];
    // Prefetch all B fragments for this K-step into registers (one clause),
    // then issue the NT WMMAs back-to-back.
    v16bf bfrag[NT];
#pragma unroll
    for (int nt = 0; nt < NT; ++nt) {
      const __bf16* bp = Bsw + ((size_t)(kt * NT + nt) * 32 + lane) * 16;
#pragma unroll
      for (int i = 0; i < 16; ++i) bfrag[nt][i] = bp[i];
    }
#pragma unroll
    for (int nt = 0; nt < NT; ++nt)
      acc[nt] = __builtin_amdgcn_wmma_f32_16x16x32_bf16(
          false, afrag, false, bfrag[nt], (short)0, acc[nt], false, false);
  }
  // Epilogue + store. C/D layout: element r -> row (lane<16 ? r : 8+r), col lane&15.
  const int rb    = (lane < 16) ? 0 : 8;
  const int mrow0 = rowBase + wave * 16;
#pragma unroll
  for (int nt = 0; nt < NT; ++nt) {
    const int col = nt * 16 + (lane & 15);
    const float bv = BIAS ? bias[col] : 0.0f;
#pragma unroll
    for (int r = 0; r < 8; ++r) {
      const int m = mrow0 + rb + r;
      float v = acc[nt][r] + bv;
      if (RES) {                                   // residual from NCHW f32 (N_==128)
        const int img = m >> 14;                   // / HW
        const int hw  = m & (HWn - 1);
        v += res_nchw[((size_t)img * Cn + col) * HWn + hw];
      }
      if (EPI == 1) v = gelu_exact(v);
      if (OBF16) ((__bf16*)outv)[(size_t)m * N_ + col] = (__bf16)v;
      else       ((float*)outv)[(size_t)m * N_ + col] = v;
    }
  }
}

// ---------------------------------------------------------------------------
// Depthwise 3x3 (pad 1) + bias + exact GELU on pixel-major bf16.
// One thread per (pixel, channel); channel fastest -> coalesced.
// ---------------------------------------------------------------------------
__global__ void dw3x3(const __bf16* __restrict__ Y1, const float* __restrict__ wd,
                      const float* __restrict__ bd, __bf16* __restrict__ Y2) {
  const long t = (long)blockIdx.x * blockDim.x + threadIdx.x;
  const int c  = (int)(t & (Cn - 1));
  const long p = t >> 7;                 // pixel = bv*HW + h*W + w
  const int hw = (int)(p & (HWn - 1));
  const int bv = (int)(p >> 14);
  const int h = hw >> 7, w = hw & (Wn - 1);
  float acc = bd[c];
#pragma unroll
  for (int dy = -1; dy <= 1; ++dy) {
    const int hh = h + dy;
    if (hh < 0 || hh >= Hn) continue;
#pragma unroll
    for (int dx = -1; dx <= 1; ++dx) {
      const int ww = w + dx;
      if (ww < 0 || ww >= Wn) continue;
      acc += wd[c * 9 + (dy + 1) * 3 + (dx + 1)] *
             (float)Y1[((size_t)bv * HWn + hh * Wn + ww) * Cn + c];
    }
  }
  Y2[(size_t)p * Cn + c] = (__bf16)gelu_exact(acc);
}

// ---------------------------------------------------------------------------
// minGRU serial scan along V (6 steps), h_{-1}=0 (matches inclusive
// associative_scan). Reads hg rows (hidden: d<128, gate: d>=128).
// One thread per (b, hw, c).
// ---------------------------------------------------------------------------
template <bool OBF16>
__global__ void gru_scan(const __bf16* __restrict__ hg, void* __restrict__ outv) {
  const long t = (long)blockIdx.x * blockDim.x + threadIdx.x;
  const int c  = (int)(t & (Cn - 1));
  const long r = t >> 7;
  const int hw = (int)(r & (HWn - 1));
  const int b  = (int)(r >> 14);
  float h = 0.0f;
#pragma unroll
  for (int v = 0; v < Vn; ++v) {
    const size_t n = (size_t)(b * Vn + v) * HWn + hw;
    const float hidden = (float)hg[n * 256 + c];
    const float gate   = (float)hg[n * 256 + 128 + c];
    const float z  = sigmoid_(gate);
    const float ht = (hidden >= 0.0f) ? (hidden + 0.5f) : sigmoid_(hidden);
    h = (1.0f - z) * h + z * ht;
    if (OBF16) ((__bf16*)outv)[n * Cn + c] = (__bf16)h;
    else       ((float*)outv)[n * Cn + c] = h;
  }
}

// ---------------------------------------------------------------------------
// alpha = 3x3 conv (C -> 1, pad 1). One wave per pixel, lanes split channels,
// wave-level shuffle reduction (wave32).
// ---------------------------------------------------------------------------
__global__ void alpha_conv(const float* __restrict__ G2f, const float* __restrict__ aw,
                           const float* __restrict__ ab, float* __restrict__ alpha) {
  const int pix  = blockIdx.x * 8 + (threadIdx.x >> 5);
  const int lane = threadIdx.x & 31;
  const int hw = pix & (HWn - 1);
  const int bv = pix >> 14;
  const int h = hw >> 7, w = hw & (Wn - 1);
  float acc = 0.0f;
  for (int dy = -1; dy <= 1; ++dy) {
    const int hh = h + dy;
    if (hh < 0 || hh >= Hn) continue;
    for (int dx = -1; dx <= 1; ++dx) {
      const int ww = w + dx;
      if (ww < 0 || ww >= Wn) continue;
      const size_t base = ((size_t)bv * HWn + hh * Wn + ww) * Cn;
      const int tap = (dy + 1) * 3 + (dx + 1);
      for (int cc = lane; cc < Cn; cc += 32) acc += aw[cc * 9 + tap] * G2f[base + cc];
    }
  }
  for (int off = 16; off > 0; off >>= 1) acc += __shfl_xor(acc, off, 32);
  if (lane == 0) alpha[pix] = acc + ab[0];
}

// ---------------------------------------------------------------------------
// Softmax over V + weighted sum -> pooled (B,HW,C) bf16 for the up-GEMM.
// ---------------------------------------------------------------------------
__global__ void view_pool(const float* __restrict__ G2f, const float* __restrict__ alpha,
                          __bf16* __restrict__ P) {
  const long t = (long)blockIdx.x * blockDim.x + threadIdx.x;
  const int c  = (int)(t & (Cn - 1));
  const long r = t >> 7;
  const int hw = (int)(r & (HWn - 1));
  const int b  = (int)(r >> 14);
  float av[Vn];
  float mx = -1e30f;
#pragma unroll
  for (int v = 0; v < Vn; ++v) {
    av[v] = alpha[(size_t)(b * Vn + v) * HWn + hw];
    mx = fmaxf(mx, av[v]);
  }
  float s = 0.0f;
#pragma unroll
  for (int v = 0; v < Vn; ++v) { av[v] = __expf(av[v] - mx); s += av[v]; }
  const float inv = 1.0f / s;
  float acc = 0.0f;
#pragma unroll
  for (int v = 0; v < Vn; ++v)
    acc += av[v] * inv * G2f[((size_t)(b * Vn + v) * HWn + hw) * Cn + c];
  P[((size_t)b * HWn + hw) * Cn + c] = (__bf16)acc;
}

// ---------------------------------------------------------------------------
// Pixel-shuffle x4 (implicit) + identity resize + 3x3 conv (8 -> 3, pad 1).
// S[b,cc,Y,X] = U[(b,Y/4,X/4), cc*16 + (Y%4)*4 + (X%4)]. One thread/output.
// ---------------------------------------------------------------------------
__global__ void outc_conv(const float* __restrict__ U, const float* __restrict__ wc,
                          const float* __restrict__ bc, float* __restrict__ out) {
  const long t = (long)blockIdx.x * blockDim.x + threadIdx.x;
  const int x = (int)(t & 511);
  const long t2 = t >> 9;
  const int y = (int)(t2 & 511);
  const long t3 = t2 >> 9;
  const int o = (int)(t3 % 3);
  const int b = (int)(t3 / 3);
  float acc = bc[o];
#pragma unroll
  for (int dy = 0; dy < 3; ++dy) {
    const int Y = y + dy - 1;
    if (Y < 0 || Y >= 512) continue;
    const int h = Y >> 2, r1 = Y & 3;
#pragma unroll
    for (int dx = 0; dx < 3; ++dx) {
      const int X = x + dx - 1;
      if (X < 0 || X >= 512) continue;
      const int w = X >> 2, r2 = X & 3;
      const size_t base = ((size_t)b * HWn + h * Wn + w) * Cn;
      const int rsub = r1 * 4 + r2;
#pragma unroll
      for (int cc = 0; cc < 8; ++cc)
        acc += wc[((o * 8 + cc) * 3 + dy) * 3 + dx] * U[base + cc * 16 + rsub];
    }
  }
  out[((size_t)(b * 3 + o) * 512 + y) * 512 + x] = acc;
}

// ---------------------------------------------------------------------------
extern "C" void kernel_launch(void* const* d_in, const int* in_sizes, int n_in,
                              void* d_out, int out_size, void* d_ws, size_t ws_size,
                              hipStream_t stream) {
  (void)in_sizes; (void)n_in; (void)out_size; (void)ws_size;
  const float* feats      = (const float*)d_in[0];
  const float* prj_feats  = (const float*)d_in[1];
  const float* merge_w1   = (const float*)d_in[2];
  const float* merge_b1   = (const float*)d_in[3];
  const float* merge_wd   = (const float*)d_in[4];
  const float* merge_bd   = (const float*)d_in[5];
  const float* merge_w2   = (const float*)d_in[6];
  const float* merge_b2   = (const float*)d_in[7];
  const float* gru_w      = (const float*)d_in[8];
  const float* gru_back_w = (const float*)d_in[9];
  const float* alpha_w    = (const float*)d_in[10];
  const float* alpha_b    = (const float*)d_in[11];
  const float* up_w       = (const float*)d_in[12];
  const float* up_b       = (const float*)d_in[13];
  const float* outc_w     = (const float*)d_in[14];
  const float* outc_b     = (const float*)d_in[15];
  // d_in[16]/d_in[17] = oh/ow = 512 -> resize is identity (hardcoded).

  char* ws = (char*)d_ws;
  // Workspace layout (peak ~362 MB), with region reuse:
  const size_t OFF_A0 = 0;                                 // NPIX*256 bf16 -> reused as G2F (NPIX*128 f32, same bytes)
  const size_t OFF_Y1 = OFF_A0 + (size_t)NPIX * 256 * 2;   // NPIX*128 bf16 -> reused as G1
  const size_t OFF_Y2 = OFF_Y1 + (size_t)NPIX * 128 * 2;   // NPIX*128 bf16 -> reused as U (BHW*128 f32)
  const size_t OFF_FM = OFF_Y2 + (size_t)NPIX * 128 * 2;   // NPIX*128 bf16
  const size_t OFF_HG = OFF_FM + (size_t)NPIX * 128 * 2;   // NPIX*256 bf16 (shared by both GRU projections)
  const size_t OFF_W1 = OFF_HG + (size_t)NPIX * 256 * 2;   // swizzled bf16 weights
  const size_t OFF_W2 = OFF_W1 + (size_t)256 * 128 * 2;
  const size_t OFF_G1W = OFF_W2 + (size_t)128 * 128 * 2;
  const size_t OFF_G2W = OFF_G1W + (size_t)128 * 256 * 2;
  const size_t OFF_UPW = OFF_G2W + (size_t)128 * 256 * 2;
  const size_t OFF_AL  = OFF_UPW + (size_t)128 * 128 * 2;  // NPIX f32
  const size_t OFF_P   = OFF_AL + (size_t)NPIX * 4;        // BHW*128 bf16

  __bf16* A0  = (__bf16*)(ws + OFF_A0);
  __bf16* Y1  = (__bf16*)(ws + OFF_Y1);
  __bf16* Y2  = (__bf16*)(ws + OFF_Y2);
  __bf16* FM  = (__bf16*)(ws + OFF_FM);
  __bf16* HG  = (__bf16*)(ws + OFF_HG);
  __bf16* G1  = (__bf16*)(ws + OFF_Y1);   // reuse
  float*  G2F = (float*)(ws + OFF_A0);    // reuse
  float*  U   = (float*)(ws + OFF_Y2);    // reuse
  __bf16* W1s = (__bf16*)(ws + OFF_W1);
  __bf16* W2s = (__bf16*)(ws + OFF_W2);
  __bf16* GW1 = (__bf16*)(ws + OFF_G1W);
  __bf16* GW2 = (__bf16*)(ws + OFF_G2W);
  __bf16* UPW = (__bf16*)(ws + OFF_UPW);
  float*  AL  = (float*)(ws + OFF_AL);
  __bf16* P   = (__bf16*)(ws + OFF_P);

  // 1) Pack activations and weights to bf16 / WMMA fragment layout.
  {
    dim3 tb(32, 8), tg(HWn / 32, Cn / 32, BVn);
    pack_act<<<tg, tb, 0, stream>>>(feats,     A0, 0,   256);
    pack_act<<<tg, tb, 0, stream>>>(prj_feats, A0, 128, 256);
  }
  pack_weight<<<(256 * 128 + 255) / 256, 256, 0, stream>>>(merge_w1,   W1s, 256, 128);
  pack_weight<<<(128 * 128 + 255) / 256, 256, 0, stream>>>(merge_w2,   W2s, 128, 128);
  pack_weight<<<(128 * 256 + 255) / 256, 256, 0, stream>>>(gru_w,      GW1, 128, 256);
  pack_weight<<<(128 * 256 + 255) / 256, 256, 0, stream>>>(gru_back_w, GW2, 128, 256);
  pack_weight<<<(128 * 128 + 255) / 256, 256, 0, stream>>>(up_w,       UPW, 128, 128);

  // 2) merge block: 1x1(2C->C)+GELU -> dw3x3+GELU -> 1x1(C->C)+residual(f)
  wmma_gemm<256, 128, 1, true, false, true>
      <<<NPIX / 128, 256, 0, stream>>>(A0, W1s, merge_b1, nullptr, Y1);
  dw3x3<<<(size_t)NPIX * Cn / 256, 256, 0, stream>>>(Y1, merge_wd, merge_bd, Y2);
  wmma_gemm<128, 128, 0, true, true, true>
      <<<NPIX / 128, 256, 0, stream>>>(Y2, W2s, merge_b2, feats, FM);

  // 3) Two forward minGRUs along V (H-flip in reference is a scan no-op).
  wmma_gemm<128, 256, 0, true, false, false>
      <<<NPIX / 128, 256, 0, stream>>>(FM, GW1, nullptr, nullptr, HG);
  gru_scan<true><<<(size_t)Bn * HWn * Cn / 256, 256, 0, stream>>>(HG, G1);
  wmma_gemm<128, 256, 0, true, false, false>
      <<<NPIX / 128, 256, 0, stream>>>(G1, GW2, nullptr, nullptr, HG);
  gru_scan<false><<<(size_t)Bn * HWn * Cn / 256, 256, 0, stream>>>(HG, G2F);

  // 4) Softmax view-attention pooling.
  alpha_conv<<<NPIX / 8, 256, 0, stream>>>(G2F, alpha_w, alpha_b, AL);
  view_pool<<<(size_t)Bn * HWn * Cn / 256, 256, 0, stream>>>(G2F, AL, P);

  // 5) up 1x1 conv (WMMA) -> pixel-shuffle x4 + identity resize + 3x3 out conv.
  wmma_gemm<128, 128, 0, false, false, true>
      <<<BHW / 128, 256, 0, stream>>>(P, UPW, up_b, nullptr, U);
  outc_conv<<<(size_t)Bn * 3 * 512 * 512 / 256, 256, 0, stream>>>(
      U, outc_w, outc_b, (float*)d_out);
}
